// ForcingLoss_65214783422667
// MI455X (gfx1250) — matrix operand order repaired and checked by reading, not compiled, not run on hardware
//
#include <hip/hip_runtime.h>
#include <hip/hip_bf16.h>

typedef __attribute__((ext_vector_type(16))) _Float16 v16h;
typedef __attribute__((ext_vector_type(8)))  _Float16 v8h;
typedef __attribute__((ext_vector_type(8)))  float    v8f;

#define HD      512
#define EPSC    1e-4f
#define AROWS   32
#define ASTRIDE 520   // 512 + 8 halfs pad: keeps 16B alignment, staggers LDS banks

// B fragment: lane's 16 consecutive K halfs for one column (w2t is [n][k] f16)
__device__ __forceinline__ v16h ldB(const _Float16* __restrict__ bb, int t, int kc) {
  return *(const v16h*)(bb + t * (16 * HD) + kc * 32);
}

// A fragment (contiguous K, matches B): halfs 0..15 -> K = kc*32+group*16+0..15
__device__ __forceinline__ v16h ldA(const _Float16* arowp, int kc, int group) {
  const int k0 = kc * 32 + group * 16;
  v8h lo = *(const v8h*)(arowp + k0);
  v8h hi = *(const v8h*)(arowp + k0 + 8);
  v16h a;
  #pragma unroll
  for (int i = 0; i < 8; ++i) { a[i] = lo[i]; a[i + 8] = hi[i]; }
  return a;
}

// ---------- prep: w2 (f32, [k][n]) -> w2t (f16, [n][k]) ----------
__global__ __launch_bounds__(256) void k_transpose_w2(const float* __restrict__ w2,
                                                      _Float16* __restrict__ w2t) {
  int idx = blockIdx.x * 256 + threadIdx.x;          // 0 .. 512*512-1
  int n = idx & (HD - 1);
  int k = idx >> 9;
  w2t[n * HD + k] = (_Float16)w2[k * HD + n];
}

// ---------- prep: build the 4x32 perturbed rows for the symplectic term ----------
__global__ void k_prep_sym(const float* __restrict__ p0, const float* __restrict__ q0,
                           float* __restrict__ pp, float* __restrict__ qq) {
  int i = threadIdx.x;                               // 128 threads
  int g = i >> 5, j = i & 31;
  float dp = (g == 0) ? EPSC : (g == 1) ? -EPSC : 0.0f;
  float dq = (g == 2) ? EPSC : (g == 3) ? -EPSC : 0.0f;
  pp[i] = p0[j] + dp;
  qq[i] = q0[j] + dq;
}

// ---------- fused 2->512->512->2 MLP with WMMA middle layer ----------
__global__ __launch_bounds__(256) void k_mlp(
    const float* __restrict__ inP, const float* __restrict__ inQ, int nrows,
    const float* __restrict__ w1, const float* __restrict__ b1,
    const _Float16* __restrict__ w2t, const float* __restrict__ b2,
    const float* __restrict__ w3, const float* __restrict__ b3,
    float* __restrict__ outP, float* __restrict__ outQ) {
  __shared__ _Float16 Alds[AROWS * ASTRIDE];
  __shared__ float opart[AROWS * 2];

  const int tid  = threadIdx.x;
  const int row0 = blockIdx.x * AROWS;

  if (tid < AROWS * 2) opart[tid] = 0.0f;

  // ---- layer 1: h1 = tanh(p*w1[0,:] + q*w1[1,:] + b1), f16 into LDS A-tile ----
  for (int idx = tid; idx < AROWS * HD; idx += 256) {
    int r = idx >> 9, j = idx & (HD - 1);
    int grow = row0 + r;
    float p = 0.0f, q = 0.0f;
    if (grow < nrows) { p = inP[grow]; q = inQ[grow]; }
    float v = tanhf(fmaf(p, w1[j], fmaf(q, w1[HD + j], b1[j])));
    Alds[r * ASTRIDE + j] = (_Float16)v;
  }
  __syncthreads();

  // ---- layer 2: 16x512 @ 512x128 slab per wave via v_wmma_f32_16x16x32_f16,
  //      with double-buffered (software-pipelined) B fragments ----
  const int wave    = tid >> 5;
  const int lane    = tid & 31;
  const int mt      = wave >> 2;            // 0..1 : which 16-row tile
  const int colbase = (wave & 3) * 128;     // 128-column slab per wave
  const int group   = lane >> 4;            // K-half selector (contiguous K order used here)
  const int m15     = lane & 15;

  const _Float16* arowp = &Alds[(mt * 16 + m15) * ASTRIDE];
  const _Float16* bb    = w2t + (size_t)(colbase + m15) * HD + group * 16;

  v8f zero = {};
  v8f acc[8];
  #pragma unroll
  for (int t = 0; t < 8; ++t) acc[t] = zero;

  v16h bA[8], bB[8];
  #pragma unroll
  for (int t = 0; t < 8; ++t) bA[t] = ldB(bb, t, 0);   // prime the pipeline

  #pragma unroll 2
  for (int kc = 0; kc < 16; kc += 2) {
    // prefetch next chunk's B while computing on bA
    #pragma unroll
    for (int t = 0; t < 8; ++t) bB[t] = ldB(bb, t, kc + 1);
    v16h a = ldA(arowp, kc, group);
    #pragma unroll
    for (int t = 0; t < 8; ++t)
      acc[t] = __builtin_amdgcn_wmma_f32_16x16x32_f16(
          false, a, false, bA[t], (short)0, acc[t], false, false);

    // prefetch chunk kc+2 while computing on bB (wave-uniform branch)
    if (kc + 2 < 16) {
      #pragma unroll
      for (int t = 0; t < 8; ++t) bA[t] = ldB(bb, t, kc + 2);
    }
    a = ldA(arowp, kc + 1, group);
    #pragma unroll
    for (int t = 0; t < 8; ++t)
      acc[t] = __builtin_amdgcn_wmma_f32_16x16x32_f16(
          false, a, false, bB[t], (short)0, acc[t], false, false);
  }

  // ---- layer 2 bias+tanh fused with layer 3 (512x2) partial dot products ----
  #pragma unroll
  for (int r = 0; r < 8; ++r) {
    float s0 = 0.0f, s1 = 0.0f;
    #pragma unroll
    for (int t = 0; t < 8; ++t) {
      const int col = colbase + t * 16 + m15;        // C/D layout: n = lane&15
      float h = tanhf(acc[t][r] + b2[col]);
      s0 = fmaf(h, w3[col * 2 + 0], s0);
      s1 = fmaf(h, w3[col * 2 + 1], s1);
    }
    const int m = mt * 16 + group * 8 + r;           // C/D layout: m = r + group*8
    atomicAdd(&opart[m * 2 + 0], s0);
    atomicAdd(&opart[m * 2 + 1], s1);
  }
  __syncthreads();

  if (tid < AROWS) {
    int grow = row0 + tid;
    if (grow < nrows) {
      outP[grow] = opart[tid * 2 + 0] + b3[0];
      outQ[grow] = opart[tid * 2 + 1] + b3[1];
    }
  }
}

// ---------- recon / action / evolution per-block partial sums ----------
__global__ __launch_bounds__(256) void k_reduce(
    const float* __restrict__ p0, const float* __restrict__ q0,
    const float* __restrict__ prec, const float* __restrict__ qrec,
    const float* __restrict__ P0, const float* __restrict__ Q0,
    const float* __restrict__ P1, const float* __restrict__ Q1,
    const float* __restrict__ Fext, const float* __restrict__ omega,
    const float* __restrict__ dt, int n, float* __restrict__ partials) {
  const float dtv = dt[0], om = omega[0];
  float rs = 0.0f, as = 0.0f, es = 0.0f;
  for (int i = blockIdx.x * blockDim.x + threadIdx.x; i < n; i += gridDim.x * blockDim.x) {
    float dp = p0[i] - prec[i];
    float dq = q0[i] - qrec[i];
    rs += dp * dp + dq * dq;
    float a = P1[i] - P0[i] - Fext[i] * dtv;
    as += a * a;
    es += 1.0f - cosf(Q1[i] - Q0[i] - om * dtv);
  }
  __shared__ float sm[3][256];
  int tid = threadIdx.x;
  sm[0][tid] = rs; sm[1][tid] = as; sm[2][tid] = es;
  __syncthreads();
  for (int off = 128; off > 0; off >>= 1) {
    if (tid < off) {
      sm[0][tid] += sm[0][tid + off];
      sm[1][tid] += sm[1][tid + off];
      sm[2][tid] += sm[2][tid + off];
    }
    __syncthreads();
  }
  if (tid == 0) {
    partials[blockIdx.x * 3 + 0] = sm[0][0];
    partials[blockIdx.x * 3 + 1] = sm[1][0];
    partials[blockIdx.x * 3 + 2] = sm[2][0];
  }
}

// ---------- symplectic loss from the 128 perturbed encodings ----------
__global__ void k_sym(const float* __restrict__ symP, const float* __restrict__ symQ,
                      float* __restrict__ symout) {
  int j = threadIdx.x;                               // 32 threads
  const float inv2e = 1.0f / (2.0f * EPSC);
  float dPdp = (symP[j]      - symP[32 + j]) * inv2e;
  float dPdq = (symP[64 + j] - symP[96 + j]) * inv2e;
  float dQdp = (symQ[j]      - symQ[32 + j]) * inv2e;
  float dQdq = (symQ[64 + j] - symQ[96 + j]) * inv2e;
  float pb = dPdq * dQdp - dPdp * dQdq;
  float v = fabsf(pb) - 1.0f;
  v *= v;
  for (int off = 16; off > 0; off >>= 1) v += __shfl_down(v, off, 32);
  if (j == 0) symout[0] = v * (1.0f / 32.0f);
}

// ---------- final deterministic combine ----------
__global__ __launch_bounds__(256) void k_combine(
    const float* __restrict__ partials, int nparts,
    const float* __restrict__ symmean, float* __restrict__ out,
    int out_size, float invB) {
  __shared__ float sm[3][256];
  int tid = threadIdx.x;
  float s0 = 0.0f, s1 = 0.0f, s2 = 0.0f;
  for (int i = tid; i < nparts; i += 256) {
    s0 += partials[i * 3 + 0];
    s1 += partials[i * 3 + 1];
    s2 += partials[i * 3 + 2];
  }
  sm[0][tid] = s0; sm[1][tid] = s1; sm[2][tid] = s2;
  __syncthreads();
  for (int off = 128; off > 0; off >>= 1) {
    if (tid < off) {
      sm[0][tid] += sm[0][tid + off];
      sm[1][tid] += sm[1][tid + off];
      sm[2][tid] += sm[2][tid + off];
    }
    __syncthreads();
  }
  if (tid == 0) {
    float recon  = sm[0][0] * invB;
    float action = sm[1][0] * invB;
    float evo    = sm[2][0] * invB;
    float sym    = symmean[0];
    float total  = 1.0f * recon + 10.0f * action + 5.0f * evo + 0.1f * sym;
    float vals[5] = { total, recon, action, evo, sym };
    for (int i = 0; i < 5 && i < out_size; ++i) out[i] = vals[i];
  }
}

extern "C" void kernel_launch(void* const* d_in, const int* in_sizes, int n_in,
                              void* d_out, int out_size, void* d_ws, size_t ws_size,
                              hipStream_t stream) {
  const float* p0     = (const float*)d_in[0];
  const float* q0     = (const float*)d_in[1];
  const float* p1     = (const float*)d_in[2];
  const float* q1     = (const float*)d_in[3];
  const float* Fext   = (const float*)d_in[4];
  const float* omega  = (const float*)d_in[5];
  const float* dt     = (const float*)d_in[6];
  const float* enc_w1 = (const float*)d_in[7];
  const float* enc_b1 = (const float*)d_in[8];
  const float* enc_w2 = (const float*)d_in[9];
  const float* enc_b2 = (const float*)d_in[10];
  const float* enc_w3 = (const float*)d_in[11];
  const float* enc_b3 = (const float*)d_in[12];
  const float* dec_w1 = (const float*)d_in[13];
  const float* dec_b1 = (const float*)d_in[14];
  const float* dec_w2 = (const float*)d_in[15];
  const float* dec_b2 = (const float*)d_in[16];
  const float* dec_w3 = (const float*)d_in[17];
  const float* dec_b3 = (const float*)d_in[18];
  const int n = in_sizes[0];                         // B = 65536

  // workspace layout
  char* ws = (char*)d_ws;
  _Float16* w2tE = (_Float16*)(ws + 0);              // 512*512 f16 = 512 KB
  _Float16* w2tD = (_Float16*)(ws + 524288);         // 512 KB
  float* P0b  = (float*)(ws + 1048576);
  float* Q0b  = P0b + n;
  float* P1b  = Q0b + n;
  float* Q1b  = P1b + n;
  float* PRb  = Q1b + n;
  float* QRb  = PRb + n;
  float* symP = QRb + n;
  float* symQ = symP + 128;
  float* pp   = symQ + 128;
  float* qq   = pp + 128;
  float* partials = qq + 128;                        // 256 blocks * 3 floats
  float* symout   = partials + 256 * 3;

  k_transpose_w2<<<(HD * HD) / 256, 256, 0, stream>>>(enc_w2, w2tE);
  k_transpose_w2<<<(HD * HD) / 256, 256, 0, stream>>>(dec_w2, w2tD);
  k_prep_sym<<<1, 128, 0, stream>>>(p0, q0, pp, qq);

  const int blocks = (n + AROWS - 1) / AROWS;        // 2048
  k_mlp<<<blocks, 256, 0, stream>>>(p0, q0, n, enc_w1, enc_b1, w2tE, enc_b2,
                                    enc_w3, enc_b3, P0b, Q0b);
  k_mlp<<<blocks, 256, 0, stream>>>(p1, q1, n, enc_w1, enc_b1, w2tE, enc_b2,
                                    enc_w3, enc_b3, P1b, Q1b);
  k_mlp<<<(128 + AROWS - 1) / AROWS, 256, 0, stream>>>(pp, qq, 128, enc_w1, enc_b1,
                                    w2tE, enc_b2, enc_w3, enc_b3, symP, symQ);
  k_mlp<<<blocks, 256, 0, stream>>>(P0b, Q0b, n, dec_w1, dec_b1, w2tD, dec_b2,
                                    dec_w3, dec_b3, PRb, QRb);

  k_reduce<<<256, 256, 0, stream>>>(p0, q0, PRb, QRb, P0b, Q0b, P1b, Q1b,
                                    Fext, omega, dt, n, partials);
  k_sym<<<1, 32, 0, stream>>>(symP, symQ, symout);
  k_combine<<<1, 256, 0, stream>>>(partials, 256, symout, (float*)d_out, out_size,
                                   1.0f / (float)n);
}
